// VisualDeconfounder_8727373546112
// MI455X (gfx1250) — matrix operand order repaired and checked by reading, not compiled, not run on hardware
//
#include <hip/hip_runtime.h>
#include <hip/hip_bf16.h>

typedef __attribute__((ext_vector_type(16))) __bf16    v16bf;
typedef __attribute__((ext_vector_type(8)))  float     v8f;
typedef __attribute__((ext_vector_type(4)))  unsigned  v4u;

union FragAB { v16bf v; v4u u[2]; };

#define TILE_M 128
#define TILE_N 128
#define TILE_K 32
#define LDS_STRIDE 48   // bf16 elems per LDS row: 32 data + 16 pad -> 96B (16B aligned)

// async global->LDS DMA, 16B per lane, tracked with ASYNCcnt
#define ASYNC_B128(ldsaddr, gptr)                                          \
    asm volatile("global_load_async_to_lds_b128 %0, %1, off"               \
                 :: "v"(ldsaddr), "v"(gptr) : "memory")

// C[M,Nn] = alpha * A[M,Kd] * B[Nn,Kd]^T + bias[Nn]; optional f32 and bf16 outputs.
__global__ __launch_bounds__(256)
void gemm_bf16_wmma(const __hip_bfloat16* __restrict__ A, int lda,
                    const __hip_bfloat16* __restrict__ B, int ldb,
                    int M, int Nn, int Kd,
                    float alpha, const float* __restrict__ bias,
                    float* __restrict__ outF, int ldcF,
                    __hip_bfloat16* __restrict__ outB, int ldcB)
{
    // [buf][A/B][tile]  -> 2 * 2 * 12KB = 48KB
    __shared__ __align__(16) unsigned short sm[2][2][TILE_M * LDS_STRIDE];

    const int tid   = threadIdx.x;
    const int lane  = tid & 31;
    const int wave  = tid >> 5;
    const int waveM = wave & 3;   // 4 waves along M, 32 rows each
    const int waveN = wave >> 2;  // 2 waves along N, 64 cols each
    const int mBase = blockIdx.y * TILE_M;
    const int nBase = blockIdx.x * TILE_N;

    // ---- loader mapping: each thread owns 4 x 16B per K-tile ----
    const int r0 = tid >> 2;            // 0..63
    const int r1 = r0 + 64;             // 64..127
    const int cg = (tid & 3) << 3;      // 0,8,16,24 (bf16 elems)

    int gra0 = mBase + r0; if (gra0 > M  - 1) gra0 = M  - 1;
    int gra1 = mBase + r1; if (gra1 > M  - 1) gra1 = M  - 1;
    int grb0 = nBase + r0; if (grb0 > Nn - 1) grb0 = Nn - 1;
    int grb1 = nBase + r1; if (grb1 > Nn - 1) grb1 = Nn - 1;

    const char* pa0 = (const char*)(A + (size_t)gra0 * lda + cg);
    const char* pa1 = (const char*)(A + (size_t)gra1 * lda + cg);
    const char* pb0 = (const char*)(B + (size_t)grb0 * ldb + cg);
    const char* pb1 = (const char*)(B + (size_t)grb1 * ldb + cg);

    // per-lane LDS destination byte addresses (low 32 bits of generic ptr = LDS offset)
    unsigned da[2][2], db[2][2];
    #pragma unroll
    for (int b = 0; b < 2; ++b) {
        da[b][0] = (unsigned)(size_t)&sm[b][0][r0 * LDS_STRIDE + cg];
        da[b][1] = (unsigned)(size_t)&sm[b][0][r1 * LDS_STRIDE + cg];
        db[b][0] = (unsigned)(size_t)&sm[b][1][r0 * LDS_STRIDE + cg];
        db[b][1] = (unsigned)(size_t)&sm[b][1][r1 * LDS_STRIDE + cg];
    }

    v8f acc[2][4] = {};
    const int lr = lane & 15;
    const int hi = lane >> 4;

    auto compute = [&](int b) {
        const unsigned short* LA = sm[b][0];
        const unsigned short* LB = sm[b][1];
        FragAB aF[2], bF[4];
        // A frag: lanes 0-15 -> K {0..7,16..23}; lanes 16-31 -> K {8..15,24..31}
        #pragma unroll
        for (int s = 0; s < 2; ++s) {
            const int row = waveM * 32 + s * 16 + lr;
            const int kb  = hi ? 8 : 0;
            aF[s].u[0] = *(const v4u*)(&LA[row * LDS_STRIDE + kb]);
            aF[s].u[1] = *(const v4u*)(&LA[row * LDS_STRIDE + kb + 16]);
        }
        // B frag: lanes 0-15 -> K {0..7,16..23}; lanes 16-31 -> K {8..15,24..31}
        #pragma unroll
        for (int t = 0; t < 4; ++t) {
            const int col = waveN * 64 + t * 16 + lr;
            const int kb  = hi ? 8 : 0;
            bF[t].u[0] = *(const v4u*)(&LB[col * LDS_STRIDE + kb]);
            bF[t].u[1] = *(const v4u*)(&LB[col * LDS_STRIDE + kb + 16]);
        }
        #pragma unroll
        for (int s = 0; s < 2; ++s)
            #pragma unroll
            for (int t = 0; t < 4; ++t)
                acc[s][t] = __builtin_amdgcn_wmma_f32_16x16x32_bf16(
                    false, aF[s].v, false, bF[t].v, (short)0, acc[s][t], false, false);
    };

    const int nIter = Kd >> 5;   // Kd is always a multiple of 32 here

    // prologue: tile 0 -> buffer 0
    ASYNC_B128(da[0][0], pa0); ASYNC_B128(da[0][1], pa1);
    ASYNC_B128(db[0][0], pb0); ASYNC_B128(db[0][1], pb1);

    int cur = 0;
    for (int it = 0; it < nIter - 1; ++it) {
        pa0 += TILE_K * 2; pa1 += TILE_K * 2;
        pb0 += TILE_K * 2; pb1 += TILE_K * 2;
        const int nxt = cur ^ 1;
        ASYNC_B128(da[nxt][0], pa0); ASYNC_B128(da[nxt][1], pa1);
        ASYNC_B128(db[nxt][0], pb0); ASYNC_B128(db[nxt][1], pb1);
        asm volatile("s_wait_asynccnt 0x4" ::: "memory");  // tile `it` landed (this wave)
        __syncthreads();                                   // ... for all waves
        compute(cur);
        __syncthreads();                                   // buffer reusable for tile it+2
        cur = nxt;
    }
    asm volatile("s_wait_asynccnt 0x0" ::: "memory");
    __syncthreads();
    compute(cur);

    // epilogue: C fragment -> VGPR r: row = r + 8*(lane>=16), col = lane%16
    const int rowAdd = hi * 8;
    #pragma unroll
    for (int s = 0; s < 2; ++s) {
        #pragma unroll
        for (int t = 0; t < 4; ++t) {
            const int gCol = nBase + waveN * 64 + t * 16 + lr;
            if (gCol >= Nn) continue;
            const float bv = bias ? bias[gCol] : 0.0f;
            #pragma unroll
            for (int r = 0; r < 8; ++r) {
                const int gRow = mBase + waveM * 32 + s * 16 + r + rowAdd;
                if (gRow >= M) continue;
                const float val = acc[s][t][r] * alpha + bv;
                if (outF) outF[(size_t)gRow * ldcF + gCol] = val;
                if (outB) outB[(size_t)gRow * ldcB + gCol] = __float2bfloat16(val);
            }
        }
    }
}

__global__ void k_f32_to_bf16(const float* __restrict__ src,
                              __hip_bfloat16* __restrict__ dst, long n) {
    long i = (long)blockIdx.x * blockDim.x + threadIdx.x;
    if (i < n) dst[i] = __float2bfloat16(src[i]);
}

// xz[r, 0:F] = bf16(x[r, :])   with xz row stride FD
__global__ void k_pack_x(const float* __restrict__ x, __hip_bfloat16* __restrict__ xz,
                         int F, int FD, long n) {
    long i = (long)blockIdx.x * blockDim.x + threadIdx.x;
    if (i < n) {
        long r = i / F, c = i - r * F;
        xz[r * FD + c] = __float2bfloat16(x[i]);
    }
}

// dicT[d, k] = bf16(dic[k, d]);  dic: [K,D] f32, dicT: [D,K] bf16
__global__ void k_transpose_bf16(const float* __restrict__ src,
                                 __hip_bfloat16* __restrict__ dst,
                                 int K, int D, long n) {
    long i = (long)blockIdx.x * blockDim.x + threadIdx.x;
    if (i < n) {
        long d = i / K, k = i - d * K;
        dst[i] = __float2bfloat16(src[k * (long)D + d]);
    }
}

// row softmax of S[N,Kd] then * prior[k] -> bf16 att_p
__global__ __launch_bounds__(256)
void k_softmax_prior(const float* __restrict__ S, const float* __restrict__ prior,
                     __hip_bfloat16* __restrict__ attp, int Kd) {
    const int row = blockIdx.x, tid = threadIdx.x;
    const float* s = S + (size_t)row * Kd;
    __shared__ float red[256];

    float m = -__builtin_inff();
    for (int k = tid; k < Kd; k += 256) m = fmaxf(m, s[k]);
    red[tid] = m; __syncthreads();
    for (int off = 128; off > 0; off >>= 1) {
        if (tid < off) red[tid] = fmaxf(red[tid], red[tid + off]);
        __syncthreads();
    }
    m = red[0]; __syncthreads();

    float sum = 0.0f;
    for (int k = tid; k < Kd; k += 256) sum += __expf(s[k] - m);
    red[tid] = sum; __syncthreads();
    for (int off = 128; off > 0; off >>= 1) {
        if (tid < off) red[tid] += red[tid + off];
        __syncthreads();
    }
    const float inv = 1.0f / red[0];

    for (int k = tid; k < Kd; k += 256)
        attp[(size_t)row * Kd + k] =
            __float2bfloat16(__expf(s[k] - m) * inv * prior[k]);
}

extern "C" void kernel_launch(void* const* d_in, const int* in_sizes, int n_in,
                              void* d_out, int out_size, void* d_ws, size_t ws_size,
                              hipStream_t stream) {
    constexpr int Nb = 8192, F = 2048, Kd = 1600, D = 1024, E = 1024, FD = F + D;
    constexpr float scale = 0.03125f; // 1/sqrt(1024)

    const float* x     = (const float*)d_in[0];
    const float* dic   = (const float*)d_in[1];
    const float* prior = (const float*)d_in[2];
    const float* Wy_w  = (const float*)d_in[3];
    const float* Wy_b  = (const float*)d_in[4];
    const float* Wz_w  = (const float*)d_in[5];
    const float* Wz_b  = (const float*)d_in[6];
    const float* cs_w  = (const float*)d_in[7];
    const float* cs_b  = (const float*)d_in[8];

    float* out_logits = (float*)d_out;                       // [N, 1600]
    float* out_z      = out_logits + (size_t)Nb * Kd;        // [N, 1024]

    // workspace carve-out (256B aligned)
    char* ws = (char*)d_ws;
    size_t off = 0;
    auto alloc = [&](size_t bytes) { char* p = ws + off; off += (bytes + 255) & ~(size_t)255; return p; };
    __hip_bfloat16* xz    = (__hip_bfloat16*)alloc((size_t)Nb * FD * 2);
    __hip_bfloat16* wyb   = (__hip_bfloat16*)alloc((size_t)E * F * 2);
    __hip_bfloat16* wzb   = (__hip_bfloat16*)alloc((size_t)E * D * 2);
    __hip_bfloat16* dicb  = (__hip_bfloat16*)alloc((size_t)Kd * D * 2);
    __hip_bfloat16* dicT  = (__hip_bfloat16*)alloc((size_t)D * Kd * 2);
    __hip_bfloat16* cswb  = (__hip_bfloat16*)alloc((size_t)Kd * FD * 2);
    __hip_bfloat16* qb    = (__hip_bfloat16*)alloc((size_t)Nb * E * 2);
    __hip_bfloat16* kbuf  = (__hip_bfloat16*)alloc((size_t)Kd * E * 2);
    float*          Sbuf  = (float*)alloc((size_t)Nb * Kd * 4);
    __hip_bfloat16* attp  = (__hip_bfloat16*)alloc((size_t)Nb * Kd * 2);
    (void)ws_size;

    auto eb = [](long n) { return dim3((unsigned)((n + 255) / 256)); };

    // precision conversion / layout
    k_pack_x<<<eb((long)Nb * F), 256, 0, stream>>>(x, xz, F, FD, (long)Nb * F);
    k_f32_to_bf16<<<eb((long)E * F), 256, 0, stream>>>(Wy_w, wyb, (long)E * F);
    k_f32_to_bf16<<<eb((long)E * D), 256, 0, stream>>>(Wz_w, wzb, (long)E * D);
    k_f32_to_bf16<<<eb((long)Kd * D), 256, 0, stream>>>(dic, dicb, (long)Kd * D);
    k_f32_to_bf16<<<eb((long)Kd * FD), 256, 0, stream>>>(cs_w, cswb, (long)Kd * FD);
    k_transpose_bf16<<<eb((long)D * Kd), 256, 0, stream>>>(dic, dicT, Kd, D, (long)D * Kd);

    auto grid = [](int M, int Nn) { return dim3((Nn + 127) / 128, (M + 127) / 128); };

    // GEMM1: q = x * Wy^T + Wy_b  -> bf16 q
    gemm_bf16_wmma<<<grid(Nb, E), 256, 0, stream>>>(
        xz, FD, wyb, F, Nb, E, F, 1.0f, Wy_b, nullptr, 0, qb, E);
    // GEMM2: k = dic * Wz^T + Wz_b -> bf16 k
    gemm_bf16_wmma<<<grid(Kd, E), 256, 0, stream>>>(
        dicb, D, wzb, D, Kd, E, D, 1.0f, Wz_b, nullptr, 0, kbuf, E);
    // GEMM3: S = scale * q * k^T -> f32
    gemm_bf16_wmma<<<grid(Nb, Kd), 256, 0, stream>>>(
        qb, E, kbuf, E, Nb, Kd, E, scale, nullptr, Sbuf, Kd, nullptr, 0);
    // softmax * prior -> bf16 att_p
    k_softmax_prior<<<dim3(Nb), 256, 0, stream>>>(Sbuf, prior, attp, Kd);
    // GEMM4: z = att_p * dicT^T -> f32 z (d_out) + bf16 into xz[:, F:]
    gemm_bf16_wmma<<<grid(Nb, D), 256, 0, stream>>>(
        attp, Kd, dicT, Kd, Nb, D, Kd, 1.0f, nullptr, out_z, D, xz + F, FD);
    // GEMM5: logits = [x,z] * cs_w^T + cs_b -> f32 (d_out)
    gemm_bf16_wmma<<<grid(Nb, Kd), 256, 0, stream>>>(
        xz, FD, cswb, FD, Nb, Kd, FD, 1.0f, cs_b, out_logits, Kd, nullptr, 0);
}
